// SA_Convlstm_cell_WithAttention_48979807043762
// MI455X (gfx1250) — matrix operand order repaired and checked by reading, not compiled, not run on hardware
//
#include <hip/hip_runtime.h>
#include <hip/hip_bf16.h>

typedef __attribute__((ext_vector_type(16))) _Float16 v16h;
typedef __attribute__((ext_vector_type(8)))  _Float16 v8h;
typedef __attribute__((ext_vector_type(8)))  float    v8f;
typedef __attribute__((ext_vector_type(4))) unsigned int u32x4;
typedef __attribute__((ext_vector_type(8))) int i32x8;
typedef __attribute__((ext_vector_type(4))) int i32x4;

#define BATCH 16
#define HID   64
#define C2    128     // CIN + HID
#define C4    256     // 4*HID
#define HH    128
#define WW    128
#define HW    16384
#define NTOK  256
#define OUTSZ 16777216ll  // 16*64*128*128

// staged conv-input tile: 3 rows x (18 cols x 128 ch) f16
#define TROW  2304            // 18*128 contiguous halfs per row
#define TPAD_STRIDE 16640     // 130*128 halfs per padded image row

__device__ __forceinline__ float sigm(float x) { return 1.f / (1.f + __expf(-x)); }
__device__ __forceinline__ int   imin(int a, int b) { return a < b ? a : b; }

__device__ __forceinline__ v16h cat8(v8h lo, v8h hi) {
  return __builtin_shufflevector(lo, hi, 0,1,2,3,4,5,6,7,8,9,10,11,12,13,14,15);
}

// ---------------------------------------------------------------- prep: pad x||h -> NHWC f16 (B,130,130,128)
__global__ void pad_kernel(const float* __restrict__ x, const float* __restrict__ h,
                           _Float16* __restrict__ xh) {
  size_t idx = (size_t)blockIdx.x * 256 + threadIdx.x;
  if (idx >= (size_t)BATCH * 130 * 130 * C2) return;
  int ch = idx & 127;
  size_t r = idx >> 7;
  int xx = (int)(r % 130); r /= 130;
  int yy = (int)(r % 130);
  int b  = (int)(r / 130);
  float v = 0.f;
  int y = yy - 1, xc = xx - 1;
  if (y >= 0 && y < HH && xc >= 0 && xc < WW) {
    if (ch < HID) v = x[((size_t)(b * HID + ch)) * HW + y * WW + xc];
    else          v = h[((size_t)(b * HID + ch - HID)) * HW + y * WW + xc];
  }
  xh[idx] = (_Float16)v;
}

// ---------------------------------------------------------------- prep: repack weights to f16
// w2[tap][oc][ic] from conv_w[oc][ic][3][3]; zw2/mw2 straight f32->f16.
__global__ void repack_kernel(const float* __restrict__ cw, const float* __restrict__ zw,
                              const float* __restrict__ mw, _Float16* __restrict__ w2,
                              _Float16* __restrict__ zw2, _Float16* __restrict__ mw2) {
  int idx = blockIdx.x * 256 + threadIdx.x;
  if (idx < 9 * 256 * 128) {
    int i = idx & 127, o = (idx >> 7) & 255, tap = idx >> 15;
    w2[idx] = (_Float16)cw[((size_t)o * 128 + i) * 9 + tap];
    return;
  }
  idx -= 9 * 256 * 128;
  if (idx < 128 * 128) { zw2[idx] = (_Float16)zw[idx]; return; }
  idx -= 128 * 128;
  if (idx < 192 * 192) { mw2[idx] = (_Float16)mw[idx]; }
}

// ---------------------------------------------------------------- conv3x3 as 9 shifted GEMMs (WMMA f16->f32)
// Block = 8 waves, 16px x 128oc; input slab staged to LDS once by the Tensor Data Mover.
// wave tile: 16 out-ch x 16 pixels; cc layout [b][p][256] f16
__global__ void conv_wmma_kernel(const _Float16* __restrict__ xh, const _Float16* __restrict__ w2,
                                 _Float16* __restrict__ cc) {
  __shared__ __align__(32) _Float16 tileB[3 * TROW];   // 13,824 B
  const int tile  = blockIdx.x;                 // 1024 pixel tiles
  const int b     = blockIdx.y;                 // 16
  const int wavei = threadIdx.x >> 5;           // 0..7
  const int octl  = blockIdx.z * 8 + wavei;     // 0..15
  const int lane  = threadIdx.x & 31;
  const int nl    = lane & 15;
  const int hi    = lane >> 4;
  const int p0    = tile * 16;
  const int y     = p0 >> 7;
  const int x0    = p0 & 127;

  const size_t gbase_h = ((size_t)(b * 130 + y) * 130 + x0) * 128;  // halfs

#if defined(__gfx1250__) && __has_builtin(__builtin_amdgcn_tensor_load_to_lds)
  // ---- TDM: one 2D descriptor load (tile 2304x3, stride 16640 elems, data_size=2B) ----
  if (wavei == 0) {
    const unsigned long long gaddr = (unsigned long long)(const void*)(xh + gbase_h);  // bytes
    const unsigned int ldsa = (unsigned int)(size_t)(void*)tileB;                      // LDS byte offset
    u32x4 g0;
    g0[0] = 1u;                                            // count=1 (valid user D#)
    g0[1] = ldsa;                                          // lds_addr
    g0[2] = (unsigned int)(gaddr & 0xffffffffu);           // global_addr[31:0]
    g0[3] = (unsigned int)((gaddr >> 32) & 0x01ffffffu)    // global_addr[56:32]
            | (2u << 30);                                  // type=2 ("image")
    i32x8 g1;
    g1[0] = (int)(1u << 16);                               // wg_mask=0, data_size=1 (2 bytes)
    g1[1] = (int)((TROW & 0xffffu) << 16);                 // tensor_dim0[15:0] -> bits 63:48
    g1[2] = (int)((TROW >> 16) | (3u << 16));              // tensor_dim0 hi | tensor_dim1[15:0]=3
    g1[3] = (int)((unsigned)TROW << 16);                   // tensor_dim1 hi=0 | tile_dim0=2304
    g1[4] = 3;                                             // tile_dim1=3, tile_dim2=0
    g1[5] = (int)TPAD_STRIDE;                              // tensor_dim0_stride lo32
    g1[6] = 0;                                             // stride hi | dim1_stride lo
    g1[7] = 0;
    i32x4 z4 = {0, 0, 0, 0};
#if __clang_major__ >= 23
    i32x8 z8 = {0, 0, 0, 0, 0, 0, 0, 0};
    __builtin_amdgcn_tensor_load_to_lds(g0, g1, z4, z4, z8, 0);
#else
    __builtin_amdgcn_tensor_load_to_lds(g0, g1, z4, z4, 0);
#endif
    __builtin_amdgcn_s_wait_tensorcnt(0);
  }
#else
  // ---- fallback: cooperative staging (864 v8h chunks) ----
  for (int j = threadIdx.x; j < 3 * TROW / 8; j += 256) {
    const int row = j / (TROW / 8), rem = j % (TROW / 8);
    ((v8h*)tileB)[j] = *(const v8h*)(xh + gbase_h + (size_t)row * TPAD_STRIDE + rem * 8);
  }
#endif
  __syncthreads();

  v8f acc = {};
  const _Float16* arow0 = w2 + (size_t)(octl * 16 + nl) * 128;
  for (int tap = 0; tap < 9; ++tap) {
    const int dy = tap / 3, dx = tap % 3;
    const _Float16* arow = arow0 + (size_t)tap * (256 * 128);
    if (tap < 8) __builtin_prefetch(arow0 + (size_t)(tap + 1) * (256 * 128), 0, 3);
    const int bofs = dy * TROW + (dx + nl) * 128 + hi * 16;
#pragma unroll
    for (int kc = 0; kc < 4; ++kc) {
      v8h alo = *(const v8h*)(arow + kc * 32 + hi * 16);
      v8h ahi = *(const v8h*)(arow + kc * 32 + hi * 16 + 8);
      v16h A  = cat8(alo, ahi);
      v16h B  = *(const v16h*)&tileB[bofs + kc * 32];
      acc = __builtin_amdgcn_wmma_f32_16x16x32_f16(false, A, false, B, (short)0, acc, false, false);
    }
  }
  // D frag: vgpr r -> (oc = octl*16 + hi*8 + r, px = p0 + nl)
  _Float16* out = cc + ((size_t)(b * HW + p0 + nl)) * 256 + octl * 16 + hi * 8;
  v8h pk;
#pragma unroll
  for (int r = 0; r < 8; ++r) pk[r] = (_Float16)acc[r];
  *(v8h*)out = pk;
}

// ---------------------------------------------------------------- GN stats: deterministic 2-stage reduction
__global__ void gn_stats_kernel(const _Float16* __restrict__ cc, float* __restrict__ psum,
                                float* __restrict__ psumsq) {
  const int chunk = blockIdx.x;   // 64 chunks of 256 pixels
  const int b = blockIdx.y;
  const int ch = threadIdx.x;     // 256
  const _Float16* base = cc + ((size_t)(b * HW + chunk * 256)) * 256 + ch;
  float s = 0.f, s2 = 0.f;
  for (int j = 0; j < 256; ++j) {
    float v = (float)base[(size_t)j * 256];
    s += v; s2 += v * v;
  }
  const int o = (chunk * 16 + b) * 256 + ch;
  psum[o] = s; psumsq[o] = s2;
}

__global__ void gn_finalize_kernel(const float* __restrict__ psum, const float* __restrict__ psumsq,
                                   const float* __restrict__ gw, const float* __restrict__ gb,
                                   float* __restrict__ ga, float* __restrict__ gb2) {
  const int i = blockIdx.x * 256 + threadIdx.x;  // 4096 = b*256+ch
  const int b = i >> 8, ch = i & 255;
  float s = 0.f, s2 = 0.f;
  for (int chunk = 0; chunk < 64; ++chunk) {
    s  += psum[(chunk * 16 + b) * 256 + ch];
    s2 += psumsq[(chunk * 16 + b) * 256 + ch];
  }
  const float mean = s * (1.f / 16384.f);
  const float var  = s2 * (1.f / 16384.f) - mean * mean;
  const float rstd = rsqrtf(var + 1e-5f);
  const float a = gw[ch] * rstd;
  ga[i]  = a;
  gb2[i] = gb[ch] - mean * a;
}

// ---------------------------------------------------------------- GN apply + LSTM gates
__global__ void gn_gate_kernel(const _Float16* __restrict__ cc, const float* __restrict__ ga,
                               const float* __restrict__ gb2, const float* __restrict__ c_in,
                               float* __restrict__ c_out, _Float16* __restrict__ hn) {
  const size_t idx = (size_t)blockIdx.x * 256 + threadIdx.x;  // NCHW flat over (b,64,HW)
  const int p = (int)(idx & 16383);
  const int c = (int)((idx >> 14) & 63);
  const int b = (int)(idx >> 20);
  const _Float16* row = cc + ((size_t)(b * HW + p)) * 256;
  const float* A  = ga  + b * 256;
  const float* B2 = gb2 + b * 256;
  const float i_ = A[c]       * (float)row[c]       + B2[c];
  const float f_ = A[64 + c]  * (float)row[64 + c]  + B2[64 + c];
  const float g_ = A[128 + c] * (float)row[128 + c] + B2[128 + c];
  const float o_ = A[192 + c] * (float)row[192 + c] + B2[192 + c];
  const float cn = sigm(f_) * c_in[idx] + sigm(i_) * tanhf(g_);
  c_out[idx] = cn;
  hn[((size_t)(b * HW + p)) * 64 + c] = (_Float16)(sigm(o_) * tanhf(cn));
}

// ---------------------------------------------------------------- pooling (64 row-major pixels per token)
__global__ void pool_h_kernel(const _Float16* __restrict__ hn, float* __restrict__ Hp) {
  const int c = blockIdx.x, b = blockIdx.y, n = threadIdx.x;
  const _Float16* base = hn + ((size_t)(b * HW + n * 64)) * 64 + c;
  float s = 0.f;
  for (int k = 0; k < 64; ++k) s += (float)base[(size_t)k * 64];
  Hp[((size_t)(b * NTOK + n)) * 64 + c] = s * (1.f / 64.f);
}

__global__ void pool_m_kernel(const float* __restrict__ m, float* __restrict__ Mp) {
  const int c = blockIdx.x, b = blockIdx.y, n = threadIdx.x;
  const float* base = m + ((size_t)(b * 64 + c)) * HW + n * 64;
  float s = 0.f;
  for (int k = 0; k < 64; ++k) s += base[k];
  Mp[((size_t)(b * NTOK + n)) * 64 + c] = s * (1.f / 64.f);
}

// ---------------------------------------------------------------- 1x1 projections on pooled tokens (tiny)
__global__ void proj_kernel(const float* __restrict__ Hp, const float* __restrict__ Mp,
                            const float* qw, const float* qb, const float* kw, const float* kb,
                            const float* vw, const float* vb, const float* kmw, const float* kmb,
                            const float* vmw, const float* vmb,
                            float* Qp, float* Kp, float* Vp, float* Kmp, float* Vmp) {
  const int n = blockIdx.x, b = blockIdx.y, pr = blockIdx.z, o = threadIdx.x;
  const float *W, *bi, *src; float* dst;
  switch (pr) {
    case 0: W = qw;  bi = qb;  src = Hp; dst = Qp;  break;
    case 1: W = kw;  bi = kb;  src = Hp; dst = Kp;  break;
    case 2: W = vw;  bi = vb;  src = Hp; dst = Vp;  break;
    case 3: W = kmw; bi = kmb; src = Mp; dst = Kmp; break;
    default:W = vmw; bi = vmb; src = Mp; dst = Vmp; break;
  }
  const float* srow = src + ((size_t)(b * NTOK + n)) * 64;
  const float* wrow = W + o * 64;
  float s = bi[o];
  for (int c = 0; c < 64; ++c) s += wrow[c] * srow[c];
  dst[((size_t)(b * NTOK + n)) * 64 + o] = s;
}

// ---------------------------------------------------------------- attention (256x256 softmax + AV), tiny
__global__ void attn_kernel(const float* __restrict__ Qp, const float* __restrict__ Kp,
                            const float* __restrict__ Vp, const float* __restrict__ Kmp,
                            const float* __restrict__ Vmp, float* __restrict__ Zh,
                            float* __restrict__ Zm) {
  const int n = blockIdx.x, b = blockIdx.y, which = blockIdx.z;
  const float* K = which ? Kmp : Kp;
  const float* V = which ? Vmp : Vp;
  float* Z = which ? Zm : Zh;
  __shared__ float qs[64];
  __shared__ float red[256];
  __shared__ float aw[256];
  __shared__ float part[4][64];
  const int t = threadIdx.x;
  if (t < 64) qs[t] = Qp[((size_t)(b * NTOK + n)) * 64 + t];
  __syncthreads();
  const float* krow = K + ((size_t)(b * NTOK + t)) * 64;
  float s = 0.f;
  for (int c = 0; c < 64; ++c) s += qs[c] * krow[c];
  red[t] = s; __syncthreads();
  for (int off = 128; off > 0; off >>= 1) {
    if (t < off) red[t] = fmaxf(red[t], red[t + off]);
    __syncthreads();
  }
  const float mx = red[0]; __syncthreads();
  const float e = __expf(s - mx);
  red[t] = e; __syncthreads();
  for (int off = 128; off > 0; off >>= 1) {
    if (t < off) red[t] += red[t + off];
    __syncthreads();
  }
  aw[t] = e / red[0];
  __syncthreads();
  const int c = t & 63, q = t >> 6;
  const float* vbase = V + ((size_t)(b * NTOK) + q * 64) * 64;
  float z = 0.f;
  for (int mm = 0; mm < 64; ++mm) z += aw[q * 64 + mm] * vbase[(size_t)mm * 64 + c];
  part[q][c] = z;
  __syncthreads();
  if (t < 64) Z[((size_t)(b * NTOK + n)) * 64 + t] = part[0][t] + part[1][t] + part[2][t] + part[3][t];
}

// ---------------------------------------------------------------- fused tail: bilinear up + z_w GEMM + m_w GEMM + gating
__global__ void final_fused_kernel(const float* __restrict__ Zh, const float* __restrict__ Zm,
                                   const _Float16* __restrict__ hn, const _Float16* __restrict__ zw2,
                                   const _Float16* __restrict__ mw2, const float* __restrict__ zb,
                                   const float* __restrict__ mb, const float* __restrict__ m_in,
                                   float* __restrict__ out_h, float* __restrict__ out_m) {
  __shared__ __align__(32) _Float16 zcat[16][128];
  __shared__ __align__(32) _Float16 comb[16][192];
  __shared__ float acc192[16][192];
  const int tile = blockIdx.x, b = blockIdx.y;
  const int t = threadIdx.x;
  const int p0 = tile * 16, y = p0 >> 7, x0 = p0 & 127;

  // stage 1: bilinear upsample 16x16 -> this row of 16 pixels (half-pixel, edge-clamp)
  float sy = (y + 0.5f) * 0.125f - 0.5f;
  sy = fmaxf(sy, 0.f);
  int y0 = imin((int)sy, 15), y1 = imin(y0 + 1, 15);
  const float wy = sy - (float)y0;
  for (int j = t; j < 2048; j += 256) {
    const int px = j & 15, ch = j >> 4;
    float sx = (x0 + px + 0.5f) * 0.125f - 0.5f;
    sx = fmaxf(sx, 0.f);
    int xx0 = imin((int)sx, 15), xx1 = imin(xx0 + 1, 15);
    const float wx = sx - (float)xx0;
    const float* base = ((ch < 64) ? Zh : Zm) + (size_t)b * NTOK * 64;
    const int c = ch & 63;
    const float v00 = base[(y0 * 16 + xx0) * 64 + c];
    const float v01 = base[(y0 * 16 + xx1) * 64 + c];
    const float v10 = base[(y1 * 16 + xx0) * 64 + c];
    const float v11 = base[(y1 * 16 + xx1) * 64 + c];
    const float v = (1.f - wy) * ((1.f - wx) * v00 + wx * v01) + wy * ((1.f - wx) * v10 + wx * v11);
    zcat[px][ch] = (_Float16)v;
  }
  // stage h_next channels into comb[:,128..191]
  for (int j = t; j < 1024; j += 256) {
    const int px = j & 15, c = j >> 4;
    comb[px][128 + c] = hn[((size_t)(b * HW + p0 + px)) * 64 + c];
  }
  __syncthreads();

  const int wavei = t >> 5, lane = t & 31, nl = lane & 15, hi = lane >> 4;
  // stage 2: Z = z_w @ zcat  (128 oc over 8 waves, K=128)
  {
    v8f acc = {};
    const _Float16* arow = zw2 + (size_t)(wavei * 16 + nl) * 128;
#pragma unroll
    for (int kc = 0; kc < 4; ++kc) {
      v8h alo = *(const v8h*)(arow + kc * 32 + hi * 16);
      v8h ahi = *(const v8h*)(arow + kc * 32 + hi * 16 + 8);
      v16h A  = cat8(alo, ahi);
      v16h B  = *(const v16h*)&zcat[nl][kc * 32 + hi * 16];
      acc = __builtin_amdgcn_wmma_f32_16x16x32_f16(false, A, false, B, (short)0, acc, false, false);
    }
    const int ocb = wavei * 16 + hi * 8;
#pragma unroll
    for (int r = 0; r < 8; ++r) comb[nl][ocb + r] = (_Float16)(acc[r] + zb[ocb + r]);
  }
  __syncthreads();
  // stage 3: combined = m_w @ [Z ; h_next]  (192 oc = 12 tiles, K=192)
  for (int tt = wavei; tt < 12; tt += 8) {
    v8f acc = {};
    const _Float16* arow = mw2 + (size_t)(tt * 16 + nl) * 192;
#pragma unroll
    for (int kc = 0; kc < 6; ++kc) {
      v8h alo = *(const v8h*)(arow + kc * 32 + hi * 16);
      v8h ahi = *(const v8h*)(arow + kc * 32 + hi * 16 + 8);
      v16h A  = cat8(alo, ahi);
      v16h B  = *(const v16h*)&comb[nl][kc * 32 + hi * 16];
      acc = __builtin_amdgcn_wmma_f32_16x16x32_f16(false, A, false, B, (short)0, acc, false, false);
    }
    const int ocb = tt * 16 + hi * 8;
#pragma unroll
    for (int r = 0; r < 8; ++r) acc192[nl][ocb + r] = acc[r] + mb[ocb + r];
  }
  __syncthreads();
  // stage 4: memory gating -> new_m, new_h
  for (int j = t; j < 1024; j += 256) {
    const int px = j & 15, c = j >> 4;
    const float mo  = acc192[px][c];
    const float mg  = acc192[px][64 + c];
    const float mi_ = acc192[px][128 + c];
    const size_t o = ((size_t)(b * 64 + c)) * HW + p0 + px;
    const float mval = m_in[o];
    const float si = sigm(mi_);
    const float nm = (1.f - si) * mval + si * tanhf(mg);
    out_m[o] = nm;
    out_h[o] = sigm(mo) * nm;
  }
}

// ---------------------------------------------------------------- host launcher
extern "C" void kernel_launch(void* const* d_in, const int* in_sizes, int n_in,
                              void* d_out, int out_size, void* d_ws, size_t ws_size,
                              hipStream_t stream) {
  (void)in_sizes; (void)n_in; (void)out_size; (void)ws_size;
  const float* x_in  = (const float*)d_in[0];
  const float* c_in  = (const float*)d_in[1];
  const float* h_in  = (const float*)d_in[2];
  const float* m_in  = (const float*)d_in[3];
  const float* convw = (const float*)d_in[4];
  // d_in[5] = conv_b: cancels exactly under per-channel GroupNorm mean removal
  const float* gn_w  = (const float*)d_in[6];
  const float* gn_b  = (const float*)d_in[7];
  const float* qh_w  = (const float*)d_in[8];  const float* qh_b = (const float*)d_in[9];
  const float* kh_w  = (const float*)d_in[10]; const float* kh_b = (const float*)d_in[11];
  const float* vh_w  = (const float*)d_in[12]; const float* vh_b = (const float*)d_in[13];
  const float* km_w  = (const float*)d_in[14]; const float* km_b = (const float*)d_in[15];
  const float* vm_w  = (const float*)d_in[16]; const float* vm_b = (const float*)d_in[17];
  const float* z_w   = (const float*)d_in[18]; const float* z_b  = (const float*)d_in[19];
  const float* m_w   = (const float*)d_in[20]; const float* m_b  = (const float*)d_in[21];

  float* out_h = (float*)d_out;
  float* out_c = (float*)d_out + OUTSZ;
  float* out_m = (float*)d_out + 2 * OUTSZ;

  char* ws = (char*)d_ws;
  size_t off = 0;
  auto take = [&](size_t bytes) { void* p = ws + off; off += (bytes + 255) & ~(size_t)255; return p; };

  _Float16* xh   = (_Float16*)take((size_t)BATCH * 130 * 130 * C2 * 2);   // 69.2 MB
  _Float16* w2   = (_Float16*)take((size_t)9 * 256 * 128 * 2);
  _Float16* zw2  = (_Float16*)take((size_t)128 * 128 * 2);
  _Float16* mw2  = (_Float16*)take((size_t)192 * 192 * 2);
  _Float16* cc   = (_Float16*)take((size_t)BATCH * HW * 256 * 2);         // 134.2 MB
  float* psum    = (float*)take((size_t)64 * 16 * 256 * 4);
  float* psumsq  = (float*)take((size_t)64 * 16 * 256 * 4);
  float* ga      = (float*)take((size_t)16 * 256 * 4);
  float* gb2     = (float*)take((size_t)16 * 256 * 4);
  _Float16* hn   = (_Float16*)take((size_t)BATCH * HW * 64 * 2);          // 33.6 MB
  float* Hp      = (float*)take((size_t)BATCH * NTOK * 64 * 4);
  float* Mp      = (float*)take((size_t)BATCH * NTOK * 64 * 4);
  float* Qp      = (float*)take((size_t)BATCH * NTOK * 64 * 4);
  float* Kp      = (float*)take((size_t)BATCH * NTOK * 64 * 4);
  float* Vp      = (float*)take((size_t)BATCH * NTOK * 64 * 4);
  float* Kmp     = (float*)take((size_t)BATCH * NTOK * 64 * 4);
  float* Vmp     = (float*)take((size_t)BATCH * NTOK * 64 * 4);
  float* Zh      = (float*)take((size_t)BATCH * NTOK * 64 * 4);
  float* Zm      = (float*)take((size_t)BATCH * NTOK * 64 * 4);

  // 1) staging / repack
  pad_kernel<<<135200, 256, 0, stream>>>(x_in, h_in, xh);
  repack_kernel<<<1360, 256, 0, stream>>>(convw, z_w, m_w, w2, zw2, mw2);
  // 2) conv3x3 via WMMA (TDM-staged input slab per block)
  conv_wmma_kernel<<<dim3(1024, 16, 2), 256, 0, stream>>>(xh, w2, cc);
  // 3) GroupNorm stats (deterministic) + gates
  gn_stats_kernel<<<dim3(64, 16), 256, 0, stream>>>(cc, psum, psumsq);
  gn_finalize_kernel<<<16, 256, 0, stream>>>(psum, psumsq, gn_w, gn_b, ga, gb2);
  gn_gate_kernel<<<65536, 256, 0, stream>>>(cc, ga, gb2, c_in, out_c, hn);
  // 4) pooling + projections + attention
  pool_h_kernel<<<dim3(64, 16), 256, 0, stream>>>(hn, Hp);
  pool_m_kernel<<<dim3(64, 16), 256, 0, stream>>>(m_in, Mp);
  proj_kernel<<<dim3(NTOK, 16, 5), 64, 0, stream>>>(Hp, Mp, qh_w, qh_b, kh_w, kh_b, vh_w, vh_b,
                                                    km_w, km_b, vm_w, vm_b, Qp, Kp, Vp, Kmp, Vmp);
  attn_kernel<<<dim3(NTOK, 16, 2), 256, 0, stream>>>(Qp, Kp, Vp, Kmp, Vmp, Zh, Zm);
  // 5) fused bilinear upsample + z_w GEMM + m_w GEMM + gating
  final_fused_kernel<<<dim3(1024, 16), 256, 0, stream>>>(Zh, Zm, hn, zw2, mw2, z_b, m_b, m_in,
                                                         out_h, out_m);
}
